// CrossAttention_21526376087832
// MI455X (gfx1250) — matrix operand order repaired and checked by reading, not compiled, not run on hardware
//
#include <hip/hip_runtime.h>

// ---------------------------------------------------------------------------
// CDNA5 (gfx1250) cross-attention, bf16 WMMA everywhere, f32 accumulation.
// ---------------------------------------------------------------------------

typedef __bf16 v16bf __attribute__((ext_vector_type(16)));
typedef __bf16 v8bf  __attribute__((ext_vector_type(8)));
typedef float  v8f   __attribute__((ext_vector_type(8)));

static __device__ inline v8f wmma_bf16(v16bf a, v16bf b, v8f c) {
    // (neg_a, A, neg_b, B, c_mod, C, reuse_a, reuse_b)
    return __builtin_amdgcn_wmma_f32_16x16x32_bf16(false, a, false, b, (short)0, c, false, false);
}

static __device__ inline v16bf cat8(v8bf lo, v8bf hi) {
    return __builtin_shufflevector(lo, hi, 0,1,2,3,4,5,6,7,8,9,10,11,12,13,14,15);
}

template <typename TX>
static __device__ inline v8bf ld8cvt(const TX* p);

template <>
__device__ inline v8bf ld8cvt<float>(const float* p) {
    v8f f = *(const v8f*)p;                 // 32B load
    return __builtin_convertvector(f, v8bf);
}
template <>
__device__ inline v8bf ld8cvt<__bf16>(const __bf16* p) {
    return *(const v8bf*)p;                 // 16B load
}

// ---------------------------------------------------------------------------
// fp32 -> bf16 conversion (weights)
// ---------------------------------------------------------------------------
__global__ void f32_to_bf16_kernel(const float* __restrict__ in,
                                   __bf16* __restrict__ out, int n) {
    int i = blockIdx.x * blockDim.x + threadIdx.x;
    if (i < n) out[i] = (__bf16)in[i];
}

// ---------------------------------------------------------------------------
// Y[M,N] = X[M,K] @ W[N,K]^T    (nn.Linear, no bias)
// Block: 256 threads = 8 waves, arranged 2(M) x 4(N). Block tile 128x128.
// Wave tile: 64x32 = 4x2 WMMA tiles. K assumed multiple of 32, N of 128.
// A-fragment (16-bit, 16x32): lanes 0-15 -> M=lane, K=0..15
//                             lanes 16-31 -> M=lane-16, K=16..31
// B-fragment (32x16):         lanes 0-15 -> N=lane, K=0..15
//                             lanes 16-31 -> N=lane-16, K=16..31
// C/D (f32 16x16):            lanes 0-15 -> N=lane, VGPR r = M=r
//                             lanes 16-31 -> N=lane-16, VGPR r = M=8+r
// ---------------------------------------------------------------------------
template <typename TX, typename TY>
__global__ __launch_bounds__(256)
void linear_wmma_kernel(const TX* __restrict__ X, const __bf16* __restrict__ W,
                        TY* __restrict__ Y, int M, int N, int K) {
    const int lane  = threadIdx.x & 31;
    const int wave  = threadIdx.x >> 5;       // 0..7
    const int waveM = wave >> 2;              // 0..1
    const int waveN = wave & 3;               // 0..3
    const int lr    = lane & 15;
    const int lh    = lane >> 4;

    const int m0 = blockIdx.x * 128 + waveM * 64;
    const int n0 = blockIdx.y * 128 + waveN * 32;

    v8f acc[4][2] = {};

    for (int k = 0; k < K; k += 32) {
        v16bf a[4], b[2];
        const v16bf zfrag = {};
#pragma unroll
        for (int i = 0; i < 4; ++i) {
            int m  = m0 + i * 16 + lr;
            int kb = k + lh * 16;
            if (m < M) {
                const TX* p = X + (size_t)m * K + kb;
                a[i] = cat8(ld8cvt<TX>(p), ld8cvt<TX>(p + 8));
            } else {
                a[i] = zfrag;
            }
        }
#pragma unroll
        for (int j = 0; j < 2; ++j) {
            int n  = n0 + j * 16 + lr;
            int kb = k + lh * 16;
            b[j] = *(const v16bf*)(W + (size_t)n * K + kb);   // 32B contiguous
        }
#pragma unroll
        for (int i = 0; i < 4; ++i)
#pragma unroll
            for (int j = 0; j < 2; ++j)
                acc[i][j] = wmma_bf16(a[i], b[j], acc[i][j]);
    }

#pragma unroll
    for (int i = 0; i < 4; ++i) {
#pragma unroll
        for (int j = 0; j < 2; ++j) {
            int n = n0 + j * 16 + lr;
#pragma unroll
            for (int r = 0; r < 8; ++r) {
                int m = m0 + i * 16 + r + lh * 8;
                if (m < M) Y[(size_t)m * N + n] = (TY)acc[i][j][r];
            }
        }
    }
}

// ---------------------------------------------------------------------------
// Pack K/V for attention:
//   Kp[b][h][t][d]  (t padded to 80, zeros beyond 76)
//   Vt[b][h][d][t]  (t padded to 96, zeros beyond 76)  -> contiguous B-frag loads
// Klin/Vlin are [B*77, 1024] with channel c = h*64 + d.
// ---------------------------------------------------------------------------
__global__ void pack_kv_kernel(const __bf16* __restrict__ Klin,
                               const __bf16* __restrict__ Vlin,
                               __bf16* __restrict__ Kp,
                               __bf16* __restrict__ Vt) {
    int idx = blockIdx.x * blockDim.x + threadIdx.x;   // over 4*16*64*96
    const int total = 4 * 16 * 64 * 96;
    if (idx >= total) return;
    int t = idx % 96;
    int d = (idx / 96) % 64;
    int h = (idx / (96 * 64)) % 16;
    int b = idx / (96 * 64 * 16);

    __bf16 zero = (__bf16)0.0f;
    size_t src = ((size_t)b * 77 + t) * 1024 + h * 64 + d;

    Vt[idx] = (t < 77) ? Vlin[src] : zero;             // Vt[b][h][d][t]
    if (t < 80) {
        __bf16 kv = (t < 77) ? Klin[src] : zero;
        Kp[(((size_t)(b * 16 + h) * 80 + t) * 64 + d)] = kv;
    }
}

// ---------------------------------------------------------------------------
// Attention: one wave per (b, h, 16-query block). 8 waves / 256-thread block.
//   S = Q K^T / 8  (5 key tiles of 16, keys padded to 80)
//   softmax over 80 cols (pads = -inf)
//   O = P V  (K-dim padded to 96), ctx[b][q][h*64+d] = O
// ---------------------------------------------------------------------------
__global__ __launch_bounds__(256)
void attn_wmma_kernel(const __bf16* __restrict__ Qlin,
                      const __bf16* __restrict__ Kp,
                      const __bf16* __restrict__ Vt,
                      __bf16* __restrict__ ctx) {
    const int lane = threadIdx.x & 31;
    const int wave = threadIdx.x >> 5;
    const int lr   = lane & 15;
    const int lh   = lane >> 4;

    int gw = blockIdx.x * 8 + wave;      // 0 .. 32767
    int qb = gw & 511;                   // 512 query blocks of 16
    int h  = (gw >> 9) & 15;
    int b  = gw >> 13;
    int q0 = qb * 16;

    __shared__ __align__(16) __bf16 plds_all[8][16 * 96];
    __bf16* plds = plds_all[wave];

    // ---- Q A-fragments (K-dim = head dim 64 -> 2 frags) ----
    const __bf16* qbase = Qlin + ((size_t)b * 8192 + q0) * 1024 + h * 64;
    v16bf qa[2];
#pragma unroll
    for (int f = 0; f < 2; ++f) {
        const __bf16* p = qbase + (size_t)lr * 1024 + f * 32 + lh * 16;
        qa[f] = cat8(*(const v8bf*)p, *(const v8bf*)(p + 8));
    }

    // ---- scores: 5 key tiles ----
    const __bf16* kpb = Kp + (size_t)(b * 16 + h) * 80 * 64;
    float s[5][8];
#pragma unroll
    for (int t = 0; t < 5; ++t) {
        v8f acc = {};
#pragma unroll
        for (int f = 0; f < 2; ++f) {
            const __bf16* p = kpb + (size_t)(t * 16 + lr) * 64 + f * 32 + lh * 16;
            acc = wmma_bf16(qa[f], *(const v16bf*)p, acc);
        }
        int tcol = t * 16 + lr;          // key index held by this lane
        bool valid = tcol < 77;
#pragma unroll
        for (int r = 0; r < 8; ++r)
            s[t][r] = valid ? acc[r] * 0.125f : -__builtin_inff();
    }

    // ---- softmax per query row (rows 0-7 in lanes 0-15, 8-15 in 16-31) ----
#pragma unroll
    for (int r = 0; r < 8; ++r) {
        float mx = s[0][r];
#pragma unroll
        for (int t = 1; t < 5; ++t) mx = fmaxf(mx, s[t][r]);
#pragma unroll
        for (int o = 8; o >= 1; o >>= 1) mx = fmaxf(mx, __shfl_xor(mx, o, 16));
        float sum = 0.0f;
#pragma unroll
        for (int t = 0; t < 5; ++t) { float e = __expf(s[t][r] - mx); s[t][r] = e; sum += e; }
#pragma unroll
        for (int o = 8; o >= 1; o >>= 1) sum += __shfl_xor(sum, o, 16);
        float inv = 1.0f / sum;
#pragma unroll
        for (int t = 0; t < 5; ++t) s[t][r] *= inv;
    }

    // ---- C-layout -> A-layout via LDS (store P as bf16, 16 x 96, zero pad) ----
#pragma unroll
    for (int t = 0; t < 5; ++t) {
        int c = t * 16 + lr;
#pragma unroll
        for (int r = 0; r < 8; ++r) {
            int m = r + lh * 8;
            plds[m * 96 + c] = (__bf16)s[t][r];
        }
    }
    {   // zero columns 80..95 (rows = lr, halves split cols 80-87 / 88-95)
        v8bf z = {};
        *(v8bf*)&plds[lr * 96 + 80 + lh * 8] = z;
    }
    __syncthreads();

    // ---- P A-fragments (K-dim 96 -> 3 frags) ----
    v16bf pa[3];
#pragma unroll
    for (int f = 0; f < 3; ++f) {
        const __bf16* p = &plds[lr * 96 + f * 32 + lh * 16];
        pa[f] = cat8(*(const v8bf*)p, *(const v8bf*)(p + 8));
    }

    // ---- O = P @ V (4 output tiles over d, 3 K-steps each) ----
    const __bf16* vtb = Vt + (size_t)(b * 16 + h) * 64 * 96;
    __bf16* cb = ctx + ((size_t)b * 8192 + q0) * 1024 + h * 64;
#pragma unroll
    for (int j = 0; j < 4; ++j) {
        v8f acc = {};
#pragma unroll
        for (int f = 0; f < 3; ++f) {
            const __bf16* p = vtb + (size_t)(j * 16 + lr) * 96 + f * 32 + lh * 16;
            acc = wmma_bf16(pa[f], *(const v16bf*)p, acc);
        }
        int d = j * 16 + lr;
#pragma unroll
        for (int r = 0; r < 8; ++r) {
            int m = r + lh * 8;
            cb[(size_t)m * 1024 + d] = (__bf16)acc[r];
        }
    }
}

// ---------------------------------------------------------------------------
// Host-side launcher
// ---------------------------------------------------------------------------
extern "C" void kernel_launch(void* const* d_in, const int* in_sizes, int n_in,
                              void* d_out, int out_size, void* d_ws, size_t ws_size,
                              hipStream_t stream) {
    constexpr int B = 4, T_V = 8192, T_T = 77, C = 1024, H = 16;
    constexpr int MQ = B * T_V;      // 32768
    constexpr int MK = B * T_T;      // 308

    const float* video = (const float*)d_in[0];
    const float* text  = (const float*)d_in[1];
    const float* Wq    = (const float*)d_in[2];
    const float* Wk    = (const float*)d_in[3];
    const float* Wv    = (const float*)d_in[4];
    const float* Wo    = (const float*)d_in[5];
    float* out = (float*)d_out;

    char* ws = (char*)d_ws;
    size_t off = 0;
    auto alloc = [&](size_t bytes) -> void* {
        void* p = ws + off;
        off += (bytes + 255) & ~(size_t)255;
        return p;
    };

    __bf16* Wq_bf = (__bf16*)alloc((size_t)C * C * 2);
    __bf16* Wk_bf = (__bf16*)alloc((size_t)C * C * 2);
    __bf16* Wv_bf = (__bf16*)alloc((size_t)C * C * 2);
    __bf16* Wo_bf = (__bf16*)alloc((size_t)C * C * 2);
    __bf16* Qlin  = (__bf16*)alloc((size_t)MQ * C * 2);
    __bf16* Klin  = (__bf16*)alloc((size_t)MK * C * 2);
    __bf16* Vlin  = (__bf16*)alloc((size_t)MK * C * 2);
    __bf16* Kp    = (__bf16*)alloc((size_t)B * H * 80 * 64 * 2);
    __bf16* Vt    = (__bf16*)alloc((size_t)B * H * 64 * 96 * 2);
    __bf16* ctx   = (__bf16*)alloc((size_t)MQ * C * 2);
    if (off > ws_size) return;   // workspace too small; bail deterministically

    // 1) weights -> bf16
    {
        int n = C * C;                         // 1,048,576
        int g = (n + 255) / 256;
        f32_to_bf16_kernel<<<g, 256, 0, stream>>>(Wq, Wq_bf, n);
        f32_to_bf16_kernel<<<g, 256, 0, stream>>>(Wk, Wk_bf, n);
        f32_to_bf16_kernel<<<g, 256, 0, stream>>>(Wv, Wv_bf, n);
        f32_to_bf16_kernel<<<g, 256, 0, stream>>>(Wo, Wo_bf, n);
    }

    // 2) projections (X is f32, converted to bf16 inside the GEMM)
    linear_wmma_kernel<float, __bf16><<<dim3(MQ / 128, C / 128), 256, 0, stream>>>(
        video, Wq_bf, Qlin, MQ, C, C);
    linear_wmma_kernel<float, __bf16><<<dim3((MK + 127) / 128, C / 128), 256, 0, stream>>>(
        text, Wk_bf, Klin, MK, C, C);
    linear_wmma_kernel<float, __bf16><<<dim3((MK + 127) / 128, C / 128), 256, 0, stream>>>(
        text, Wv_bf, Vlin, MK, C, C);

    // 3) pack K (padded) and V (transposed, padded)
    pack_kv_kernel<<<(B * H * 64 * 96) / 256, 256, 0, stream>>>(Klin, Vlin, Kp, Vt);

    // 4) attention -> ctx
    attn_wmma_kernel<<<(B * H * (T_V / 16)) / 8, 256, 0, stream>>>(Qlin, Kp, Vt, ctx);

    // 5) output projection -> f32 result
    linear_wmma_kernel<__bf16, float><<<dim3(MQ / 128, C / 128), 256, 0, stream>>>(
        ctx, Wo_bf, out, MQ, C, C);
    (void)in_sizes; (void)n_in; (void)out_size;
}
